// MSPAgent_80066780332773
// MI455X (gfx1250) — matrix operand order repaired and checked by reading, not compiled, not run on hardware
//
#include <hip/hip_runtime.h>
#include <hip/hip_bf16.h>

#define DEVINL __device__ __forceinline__

typedef _Float16 h16;
typedef __attribute__((ext_vector_type(16))) _Float16 v16h;
typedef __attribute__((ext_vector_type(4)))  _Float16 h4;
typedef __attribute__((ext_vector_type(2)))  _Float16 h2;
typedef __attribute__((ext_vector_type(8)))  float    v8f;

// ---------------- problem constants ----------------
constexpr int NROW    = 16384 * 8;   // M = B*A = 131072
constexpr int EDIM    = 128;
constexpr int HDIM    = 64;
constexpr int NEXPERT = 8;
constexpr int NACTION = 32;
constexpr int NTILES  = NROW / 16;   // 8192 16-row tiles
constexpr int NWAVE   = 8;           // waves per workgroup
constexpr int BLOCK   = NWAVE * 32;  // 256 threads
constexpr int NWG     = 256;         // workgroups: 4 tiles per wave, amortizes
                                     // the LDS weight staging 4x (L2 traffic)

// padded LDS strides (in halfs) so 16-row fragment gathers hit distinct banks
constexpr int SP_E = 132;            // for K=128 matrices
constexpr int SP_H = 68;             // for K=64 matrices

// ---------------- LDS layout (byte offsets) ----------------
constexpr size_t OFF_FCW = 0;
constexpr size_t SZ_FCW  = 64  * SP_E * 2;          // fc_w   [64][128]  f16
constexpr size_t OFF_WIH = OFF_FCW + SZ_FCW;
constexpr size_t SZ_WIH  = 192 * SP_H * 2;          // w_ih   [192][64]  f16
constexpr size_t OFF_WHH = OFF_WIH + SZ_WIH;
constexpr size_t SZ_WHH  = SZ_WIH;                  // w_hh
constexpr size_t OFF_EW1 = OFF_WHH + SZ_WHH;
constexpr size_t SZ_EW1  = 8 * 64 * SP_H * 2;       // e_w1 [8][64][64]  f16
constexpr size_t OFF_EW2 = OFF_EW1 + SZ_EW1;
constexpr size_t SZ_EW2  = 8 * 32 * SP_H * 2;       // e_w2 [8][32][64]  f16
constexpr size_t OFF_GW  = OFF_EW2 + SZ_EW2;
constexpr size_t SZ_GW   = 16 * SP_E * 2;           // gate_w padded [16][128] f16
constexpr size_t OFF_FCB = OFF_GW  + SZ_GW;         // f32 biases below
constexpr size_t OFF_BIH = OFF_FCB + 64  * 4;
constexpr size_t OFF_BHH = OFF_BIH + 192 * 4;
constexpr size_t OFF_EB1 = OFF_BHH + 192 * 4;
constexpr size_t OFF_EB2 = OFF_EB1 + 512 * 4;
constexpr size_t OFF_GB  = OFF_EB2 + 256 * 4;
constexpr size_t OFF_WV  = OFF_GB  + 16  * 4;       // start of per-wave scratch
// per-wave scratch (byte offsets within a wave's slice)
constexpr size_t WOFF_XIN = 0;                       // x_in tile [16][SP_E] f16
constexpr size_t WOFF_HIN = WOFF_XIN + 16 * SP_E * 2;// h_in tile [16][SP_H] f16
constexpr size_t WOFF_XR  = WOFF_HIN + 16 * SP_H * 2;// relu(fc)  [16][SP_H] f16
constexpr size_t WOFF_HH  = WOFF_XR  + 16 * SP_H * 2;// h (GRU)   [16][SP_H] f16
constexpr size_t WOFF_H1  = WOFF_HH  + 16 * SP_H * 2;// expert h1 [16][SP_H] f16
constexpr size_t WOFF_WGT = WOFF_H1  + 16 * SP_H * 2;// gate vals [16][8]    f32
constexpr size_t WAVE_SZ  = WOFF_WGT + 16 * 8 * 4;   // 13440 B
constexpr size_t SMEM_SZ  = OFF_WV + (size_t)NWAVE * WAVE_SZ; // 290240 B < 320 KB

// ---------------- helpers ----------------
DEVINL void lds_fence() { asm volatile("s_wait_dscnt 0" ::: "memory"); }

DEVINL v8f wmma_f16(v16h a, v16h b, v8f c) {
  // D(16x16 f32) = A(16x32 f16) * B(32x16 f16) + C
  return __builtin_amdgcn_wmma_f32_16x16x32_f16(false, a, false, b, (short)0, c,
                                                false, false);
}

DEVINL v8f splat8(float x) {
  v8f r;
#pragma unroll
  for (int i = 0; i < 8; ++i) r[i] = x;
  return r;
}

DEVINL float fsigmoid(float x) { return 1.0f / (1.0f + __expf(-x)); }

// A fragment (16x32 f16): lane holds row M=lane%16, K per ISA layout:
//   K = 16*(v/4) + 8*(lane/16) + 2*(v%4) + slot   (K,K+1 contiguous pairs)
DEVINL v16h load_a(const h16* t, int stride, int k0, int mrow, int hsel) {
  v16h a;
  const h16* base = t + mrow * stride + k0 + (hsel << 3);
#pragma unroll
  for (int v = 0; v < 8; ++v) {
    const int k = ((v >> 2) << 4) + ((v & 3) << 1);
    h2 p = *(const h2*)(base + k);
    a[2 * v]     = p[0];
    a[2 * v + 1] = p[1];
  }
  return a;
}

// B fragment (32x16 f16) from [N][K] row-major weights: lane holds col
// N=n0+lane%16; K = k0 + 16*(v/4) + 8*(lane/16) + 2*(v%4) + slot (same as A).
DEVINL v16h load_b(const h16* t, int stride, int n0, int k0, int lane) {
  v16h b;
  const h16* base = t + (n0 + (lane & 15)) * stride + k0 + ((lane >> 4) << 3);
#pragma unroll
  for (int v = 0; v < 8; ++v) {
    h2 p = *(const h2*)(base + ((v >> 2) << 4) + ((v & 3) << 1));
    b[2 * v]     = p[0];
    b[2 * v + 1] = p[1];
  }
  return b;
}

// ---------------- kernel ----------------
__global__ __launch_bounds__(BLOCK) void msp_agent_kernel(
    const float* __restrict__ x_in,  const float* __restrict__ h_in,
    const float* __restrict__ fc_w,  const float* __restrict__ fc_b,
    const float* __restrict__ w_ih,  const float* __restrict__ w_hh,
    const float* __restrict__ b_ih,  const float* __restrict__ b_hh,
    const float* __restrict__ e_w1,  const float* __restrict__ e_b1,
    const float* __restrict__ e_w2,  const float* __restrict__ e_b2,
    const float* __restrict__ gate_w,const float* __restrict__ gate_b,
    float* __restrict__ out_fq, float* __restrict__ out_h,
    float* __restrict__ out_eo)
{
  extern __shared__ char smem[];
  h16*   sFCW = (h16*)(smem + OFF_FCW);
  h16*   sWIH = (h16*)(smem + OFF_WIH);
  h16*   sWHH = (h16*)(smem + OFF_WHH);
  h16*   sEW1 = (h16*)(smem + OFF_EW1);
  h16*   sEW2 = (h16*)(smem + OFF_EW2);
  h16*   sGW  = (h16*)(smem + OFF_GW);
  float* sFCB = (float*)(smem + OFF_FCB);
  float* sBIH = (float*)(smem + OFF_BIH);
  float* sBHH = (float*)(smem + OFF_BHH);
  float* sEB1 = (float*)(smem + OFF_EB1);
  float* sEB2 = (float*)(smem + OFF_EB2);
  float* sGB  = (float*)(smem + OFF_GB);

  const int tid  = threadIdx.x;
  const int wave = tid >> 5;
  const int lane = tid & 31;
  const int mrow = lane & 15;   // row within A tile / col within B and D tiles
  const int hsel = lane >> 4;   // lane-half select

  // ---- cooperative weight conversion f32 -> f16 into LDS (padded strides)
  for (int i = tid; i < 64 * 128;  i += BLOCK) sFCW[(i >> 7) * SP_E + (i & 127)] = (h16)fc_w[i];
  for (int i = tid; i < 192 * 64;  i += BLOCK) sWIH[(i >> 6) * SP_H + (i & 63)]  = (h16)w_ih[i];
  for (int i = tid; i < 192 * 64;  i += BLOCK) sWHH[(i >> 6) * SP_H + (i & 63)]  = (h16)w_hh[i];
  for (int i = tid; i < 8 * 64 * 64; i += BLOCK) sEW1[(i >> 6) * SP_H + (i & 63)] = (h16)e_w1[i];
  for (int i = tid; i < 8 * 32 * 64; i += BLOCK) sEW2[(i >> 6) * SP_H + (i & 63)] = (h16)e_w2[i];
  for (int i = tid; i < 16 * 128; i += BLOCK)
    sGW[(i >> 7) * SP_E + (i & 127)] = ((i >> 7) < 8) ? (h16)gate_w[i] : (h16)0.0f;
  for (int i = tid; i < 64;  i += BLOCK) sFCB[i] = fc_b[i];
  for (int i = tid; i < 192; i += BLOCK) { sBIH[i] = b_ih[i]; sBHH[i] = b_hh[i]; }
  for (int i = tid; i < 512; i += BLOCK) sEB1[i] = e_b1[i];
  for (int i = tid; i < 256; i += BLOCK) sEB2[i] = e_b2[i];
  if (tid < 16) sGB[tid] = (tid < 8) ? gate_b[tid] : 0.0f;
  __syncthreads();

  char* wbase = smem + OFF_WV + (size_t)wave * WAVE_SZ;
  h16*   sXIN = (h16*)(wbase + WOFF_XIN);
  h16*   sHIN = (h16*)(wbase + WOFF_HIN);
  h16*   sXR  = (h16*)(wbase + WOFF_XR);
  h16*   sHH  = (h16*)(wbase + WOFF_HH);
  h16*   sH1  = (h16*)(wbase + WOFF_H1);
  float* sWGT = (float*)(wbase + WOFF_WGT);

  for (int tile = blockIdx.x * NWAVE + wave; tile < NTILES;
       tile += gridDim.x * NWAVE) {
    const int r0 = tile * 16;

    // ---- stage x_in [16][128] and h_in [16][64] tiles as f16 in LDS
    {
      const float* src = x_in + (size_t)(r0 + mrow) * EDIM + hsel * 64;
      h16* dst = sXIN + mrow * SP_E + hsel * 64;
#pragma unroll
      for (int c = 0; c < 64; c += 4) {
        float4 f = *(const float4*)(src + c);
        h4 hv; hv[0] = (h16)f.x; hv[1] = (h16)f.y; hv[2] = (h16)f.z; hv[3] = (h16)f.w;
        *(h4*)(dst + c) = hv;
      }
      const float* srch = h_in + (size_t)(r0 + mrow) * HDIM + hsel * 32;
      h16* dsth = sHIN + mrow * SP_H + hsel * 32;
#pragma unroll
      for (int c = 0; c < 32; c += 4) {
        float4 f = *(const float4*)(srch + c);
        h4 hv; hv[0] = (h16)f.x; hv[1] = (h16)f.y; hv[2] = (h16)f.z; hv[3] = (h16)f.w;
        *(h4*)(dsth + c) = hv;
      }
    }
    lds_fence();

    // ---- fc + gate share the x_in A fragments (hoisted: loaded once)
    {
      v16h aE[4];
#pragma unroll
      for (int kt = 0; kt < 4; ++kt)
        aE[kt] = load_a(sXIN, SP_E, kt * 32, mrow, hsel);

      // fc: x = relu(x_in @ fc_w^T + fc_b)   (4 N-tiles x 4 K-steps)
#pragma unroll
      for (int nt = 0; nt < 4; ++nt) {
        const int n0 = nt * 16;
        v8f acc = splat8(sFCB[n0 + mrow]);
#pragma unroll
        for (int kt = 0; kt < 4; ++kt)
          acc = wmma_f16(aE[kt], load_b(sFCW, SP_E, n0, kt * 32, lane), acc);
#pragma unroll
        for (int v = 0; v < 8; ++v) {
          const float xv = acc[v] > 0.0f ? acc[v] : 0.0f;
          sXR[(hsel * 8 + v) * SP_H + n0 + mrow] = (h16)xv;
        }
      }

      // gate: weight = sigmoid(x_in @ gate_w^T + gate_b)  (N=8, padded to 16)
      {
        v8f acc = splat8(sGB[mrow]);
#pragma unroll
        for (int kt = 0; kt < 4; ++kt)
          acc = wmma_f16(aE[kt], load_b(sGW, SP_E, 0, kt * 32, lane), acc);
        if (mrow < 8) {
#pragma unroll
          for (int v = 0; v < 8; ++v)
            sWGT[(hsel * 8 + v) * 8 + mrow] = fsigmoid(acc[v]);
        }
      }
    }
    lds_fence();

    // ---- GRU: hoist aX/aH fragments, r/z/n fused per 16-col j-tile
    {
      v16h aX[2], aH[2];
#pragma unroll
      for (int kt = 0; kt < 2; ++kt) {
        aX[kt] = load_a(sXR,  SP_H, kt * 32, mrow, hsel);
        aH[kt] = load_a(sHIN, SP_H, kt * 32, mrow, hsel);
      }
#pragma unroll
      for (int jt = 0; jt < 4; ++jt) {
        const int j0 = jt * 16;
        v8f aRi = splat8(sBIH[j0 + mrow]);
        v8f aZi = splat8(sBIH[64 + j0 + mrow]);
        v8f aNi = splat8(sBIH[128 + j0 + mrow]);
        v8f aRh = splat8(sBHH[j0 + mrow]);
        v8f aZh = splat8(sBHH[64 + j0 + mrow]);
        v8f aNh = splat8(sBHH[128 + j0 + mrow]);
#pragma unroll
        for (int kt = 0; kt < 2; ++kt) {
          const int k0 = kt * 32;
          aRi = wmma_f16(aX[kt], load_b(sWIH, SP_H, j0,       k0, lane), aRi);
          aZi = wmma_f16(aX[kt], load_b(sWIH, SP_H, 64 + j0,  k0, lane), aZi);
          aNi = wmma_f16(aX[kt], load_b(sWIH, SP_H, 128 + j0, k0, lane), aNi);
          aRh = wmma_f16(aH[kt], load_b(sWHH, SP_H, j0,       k0, lane), aRh);
          aZh = wmma_f16(aH[kt], load_b(sWHH, SP_H, 64 + j0,  k0, lane), aZh);
          aNh = wmma_f16(aH[kt], load_b(sWHH, SP_H, 128 + j0, k0, lane), aNh);
        }
#pragma unroll
        for (int v = 0; v < 8; ++v) {
          const int m = hsel * 8 + v;
          const float r = fsigmoid(aRi[v] + aRh[v]);
          const float z = fsigmoid(aZi[v] + aZh[v]);
          const float n = tanhf(aNi[v] + r * aNh[v]);
          const float hprev = (float)sHIN[m * SP_H + j0 + mrow];
          const float h = (1.0f - z) * n + z * hprev;
          out_h[(size_t)(r0 + m) * HDIM + j0 + mrow] = h;
          sHH[m * SP_H + j0 + mrow] = (h16)h;
        }
      }
    }
    lds_fence();

    // ---- experts + weighted mean (h fragments hoisted across all experts)
    v8f fq0 = splat8(0.0f);
    v8f fq1 = splat8(0.0f);
    {
      v16h aHH[2];
#pragma unroll
      for (int kt = 0; kt < 2; ++kt)
        aHH[kt] = load_a(sHH, SP_H, kt * 32, mrow, hsel);

      for (int n = 0; n < NEXPERT; ++n) {
        const h16* w1 = sEW1 + n * 64 * SP_H;
        const h16* w2 = sEW2 + n * 32 * SP_H;
        // h1 = relu(h @ e_w1[n]^T + e_b1[n])
#pragma unroll
        for (int jt = 0; jt < 4; ++jt) {
          const int j0 = jt * 16;
          v8f acc = splat8(sEB1[n * 64 + j0 + mrow]);
#pragma unroll
          for (int kt = 0; kt < 2; ++kt)
            acc = wmma_f16(aHH[kt], load_b(w1, SP_H, j0, kt * 32, lane), acc);
#pragma unroll
          for (int v = 0; v < 8; ++v) {
            const float xv = acc[v] > 0.0f ? acc[v] : 0.0f;
            sH1[(hsel * 8 + v) * SP_H + j0 + mrow] = (h16)xv;
          }
        }
        lds_fence();
        // per-lane gate weights + h1 fragments for this expert
        float wg[8];
#pragma unroll
        for (int v = 0; v < 8; ++v) wg[v] = sWGT[(hsel * 8 + v) * 8 + n];
        v16h aH1[2];
#pragma unroll
        for (int kt = 0; kt < 2; ++kt)
          aH1[kt] = load_a(sH1, SP_H, kt * 32, mrow, hsel);
        // eo = h1 @ e_w2[n]^T + e_b2[n]  (2 action tiles)
#pragma unroll
        for (int at = 0; at < 2; ++at) {
          const int a0 = at * 16;
          v8f acc = splat8(sEB2[n * 32 + a0 + mrow]);
#pragma unroll
          for (int kt = 0; kt < 2; ++kt)
            acc = wmma_f16(aH1[kt], load_b(w2, SP_H, a0, kt * 32, lane), acc);
#pragma unroll
          for (int v = 0; v < 8; ++v) {
            const int m = hsel * 8 + v;
            const float eo = acc[v];
            out_eo[((size_t)(r0 + m) * NEXPERT + n) * NACTION + a0 + mrow] = eo;
            if (at == 0) fq0[v] += eo * wg[v]; else fq1[v] += eo * wg[v];
          }
        }
        lds_fence();
      }
    }
#pragma unroll
    for (int v = 0; v < 8; ++v) {
      const int m = hsel * 8 + v;
      out_fq[(size_t)(r0 + m) * NACTION + mrow]      = fq0[v] * 0.125f;
      out_fq[(size_t)(r0 + m) * NACTION + 16 + mrow] = fq1[v] * 0.125f;
    }
  }
}

// ---------------- host launcher ----------------
extern "C" void kernel_launch(void* const* d_in, const int* in_sizes, int n_in,
                              void* d_out, int out_size, void* d_ws, size_t ws_size,
                              hipStream_t stream) {
  (void)in_sizes; (void)n_in; (void)out_size; (void)d_ws; (void)ws_size;
  const float* x_in   = (const float*)d_in[0];
  const float* h_in   = (const float*)d_in[1];
  const float* fc_w   = (const float*)d_in[2];
  const float* fc_b   = (const float*)d_in[3];
  const float* w_ih   = (const float*)d_in[4];
  const float* w_hh   = (const float*)d_in[5];
  const float* b_ih   = (const float*)d_in[6];
  const float* b_hh   = (const float*)d_in[7];
  const float* e_w1   = (const float*)d_in[8];
  const float* e_b1   = (const float*)d_in[9];
  const float* e_w2   = (const float*)d_in[10];
  const float* e_b2   = (const float*)d_in[11];
  const float* gate_w = (const float*)d_in[12];
  const float* gate_b = (const float*)d_in[13];

  float* out    = (float*)d_out;
  float* out_fq = out;                                   // [M,32]
  float* out_h  = out + (size_t)NROW * NACTION;          // [M,64]
  float* out_eo = out + (size_t)NROW * (NACTION + HDIM); // [M,8,32]

  dim3 grid(NWG);              // 256 workgroups, 4 tiles per wave
  dim3 block(BLOCK);           // 256 threads = 8 wave32
  msp_agent_kernel<<<grid, block, SMEM_SZ, stream>>>(
      x_in, h_in, fc_w, fc_b, w_ih, w_hh, b_ih, b_hh,
      e_w1, e_b1, e_w2, e_b2, gate_w, gate_b,
      out_fq, out_h, out_eo);
}
